// Categorical2DSemanticMapModule_50259707298620
// MI455X (gfx1250) — matrix operand order repaired and checked by reading, not compiled, not run on hardware
//
#include <hip/hip_runtime.h>
#include <hip/hip_bf16.h>
#include <math.h>

typedef __attribute__((ext_vector_type(16))) _Float16 v16h;
typedef __attribute__((ext_vector_type(8)))  float    v8f;

// TDM descriptor vector types (clang-23 / therock-10.0 6-arg builtin form)
typedef unsigned int tdm_v4u __attribute__((ext_vector_type(4)));
typedef int          tdm_v8i __attribute__((ext_vector_type(8)));
typedef int          tdm_v4i __attribute__((ext_vector_type(4)));

#if defined(__gfx1250__) && __has_builtin(__builtin_amdgcn_tensor_load_to_lds) && __has_builtin(__builtin_amdgcn_s_wait_tensorcnt)
#define USE_TDM 1
#else
#define USE_TDM 0
#endif

#define BB 2
#define TT 4
#define NSEM 16
#define CMAP 20
#define LOCN 480
#define GLON 960
#define VRN 100
#define ZBN 80
#define PH 120
#define PW 160
#define NPTS (PH*PW)
#define FC 17
#define GVOX (VRN*VRN*ZBN)
#define FRW 640
#define FRH 480
#define MINZ 13
#define MAXZ 25
#define DEGC 57.29577951308232f
#define PIF 3.14159265358979323846f

// ---------------- workspace layout (float offsets) ----------------
#define WS_LMAP  ((size_t)0)
#define WS_GMAP  (WS_LMAP + (size_t)BB*CMAP*LOCN*LOCN)
#define WS_AV    (WS_GMAP + (size_t)BB*CMAP*GLON*GLON)
#define WS_ROT   (WS_AV   + (size_t)BB*CMAP*LOCN*LOCN)
#define WS_VOX   (WS_ROT  + (size_t)BB*CMAP*LOCN*LOCN)
#define WS_FEAT  (WS_VOX  + (size_t)BB*FC*GVOX)
#define WS_COORD (WS_FEAT + (size_t)BB*FC*NPTS)
#define WS_TMP0  (WS_COORD+ (size_t)BB*3*NPTS)
#define WS_ROWS  (WS_TMP0 + (size_t)BB*LOCN*LOCN)                 // _Float16 area
#define WS_POSE  (WS_ROWS + ((size_t)BB*NSEM*PH*FRW + 1)/2)       // 18 floats used
#define WS_LMB   (WS_POSE + 24)                                   // 8 ints

// ---------------- output layout (float offsets) ----------------
#define OFF_FE   ((size_t)0)
#define OFF_LMAP (OFF_FE   + (size_t)BB*TT*24*LOCN*LOCN)
#define OFF_GMAP (OFF_LMAP + (size_t)BB*CMAP*LOCN*LOCN)
#define OFF_LP   (OFF_GMAP + (size_t)BB*CMAP*GLON*GLON)
#define OFF_GP   (OFF_LP + (size_t)BB*TT*3)
#define OFF_LB   (OFF_GP + (size_t)BB*TT*3)
#define OFF_OG   (OFF_LB + (size_t)BB*TT*4)

__global__ void k_zero(float* p, size_t n) {
  size_t i = (size_t)blockIdx.x * blockDim.x + threadIdx.x;
  size_t stride = (size_t)gridDim.x * blockDim.x;
  for (; i < n; i += stride) p[i] = 0.f;
}

__global__ void k_apply_done_map(float* map, size_t per_b, const unsigned char* dones, int t) {
  size_t i = (size_t)blockIdx.x * blockDim.x + threadIdx.x;
  if (i >= (size_t)BB * per_b) return;
  int b = (int)(i / per_b);
  if (dones[b*TT + t]) map[i] = 0.f;
}

__global__ void k_reset_pose(float* pose, int* lmb, const unsigned char* dones, int t) {
  int b = threadIdx.x; if (b >= BB) return;
  if (!dones[b*TT + t]) return;
  float* lp = pose + b*3;       lp[0]=12.f; lp[1]=12.f; lp[2]=0.f;
  float* gp = pose + 6 + b*3;   gp[0]=24.f; gp[1]=24.f; gp[2]=0.f;
  float* og = pose + 12 + b*3;  og[0]=12.f; og[1]=12.f; og[2]=0.f;
  int* l = lmb + b*4; l[0]=240; l[1]=720; l[2]=240; l[3]=720;
}

__global__ void k_new_pose(float* pose, const float* pdelta, int t) {
  int b = threadIdx.x; if (b >= BB) return;
  float* lp = pose + b*3;
  const float* r = pdelta + ((size_t)b*TT + t)*3;
  float s = sinf(lp[2]/DEGC), c = cosf(lp[2]/DEGC);
  float x = lp[0] + r[0]*c - r[1]*s;
  float y = lp[1] + r[0]*s + r[1]*c;
  float o = lp[2] + r[2]*DEGC;
  o = fmodf(o - 180.f, 360.f) + 180.f;
  o = fmodf(o + 180.f, 360.f) - 180.f;
  lp[0]=x; lp[1]=y; lp[2]=o;
}

// vertical 4-row sums of seg channels -> f16 (values 0..4, exact in f16)
__global__ void k_rowsum(const float* obs, _Float16* rows, int t) {
  size_t i = (size_t)blockIdx.x * blockDim.x + threadIdx.x;
  size_t n = (size_t)BB*NSEM*PH*FRW;
  if (i >= n) return;
  int k = (int)(i % FRW); size_t r = i / FRW;
  int y = (int)(r % PH); r /= PH;
  int c = (int)(r % NSEM); int b = (int)(r / NSEM);
  const float* src = obs + ((((size_t)b*TT + t)*CMAP + 4 + c)*FRH + 4*(size_t)y)*FRW + k;
  float s = src[0] + src[(size_t)FRW] + src[2*(size_t)FRW] + src[3*(size_t)FRW];
  rows[i] = (_Float16)s;
}

// Horizontal pooling contraction: out[m, x] = sum_k rowsum[m,k] * Q[k,x], Q[k,x]=1/16 iff k/4==x.
// Per wave: one 16x16 output tile = exactly one 16x64 contiguous-K A tile -> staged to LDS by the
// Tensor Data Mover (2D D#), then consumed by two v_wmma_f32_16x16x32_f16.
__global__ void k_pool_wmma(const _Float16* __restrict__ rows, float* __restrict__ feat) {
#if USE_TDM
  __shared__ _Float16 smem[8 * 16 * 64];          // 8 waves/block * 2KB tile
#endif
  int gtid = blockIdx.x * blockDim.x + threadIdx.x;
  int wave = gtid >> 5, lane = gtid & 31;
  int tn = wave % 10; int w2 = wave / 10;
  int tm = w2 % 120;  int b  = w2 / 120;          // exact sizing: b < BB always
  int xbase = tn*16;
  int xg    = xbase + (lane & 15);
  int kstart = xbase*4;                           // 64 contiguous K per tile
  int hiA = (lane >= 16) ? 8  : 0;                // A 16x32 f16 lane/K layout (ISA 7.12.2)
  int hiB = (lane >= 16) ? 16 : 0;                // B 32x16: same K order as A (table order)

#if USE_TDM
  int wib = threadIdx.x >> 5;                     // wave index in block
  unsigned slab = (unsigned)(wib * 16 * 64 * 2);  // LDS byte offset of this wave's tile
  unsigned long long gaddr =
      (unsigned long long)(uintptr_t)(rows + ((size_t)b*1920 + (size_t)tm*16)*FRW + kstart);
  // D# group0: count=1 | lds_addr | global_addr(57b) | type=2  (ISA 8.3)
  tdm_v4u g0;
  g0[0] = 1u;
  g0[1] = slab;
  g0[2] = (unsigned)(gaddr & 0xFFFFFFFFull);
  g0[3] = (unsigned)((gaddr >> 32) & 0x01FFFFFFull) | (2u << 30);
  // D# group1: data_size=2B; tensor_dim0=640, tensor_dim1=16; tile 64x16; dim0_stride=640 (ISA 8.4)
  tdm_v8i g1;
  g1[0] = (int)(1u << 16);          // data_size = 1 -> 2 bytes
  g1[1] = (int)(640u << 16);        // tensor_dim0[15:0] @ bits 63:48
  g1[2] = (int)(16u << 16);         // tensor_dim0[31:16]=0 | tensor_dim1[15:0] @ bits 95:80
  g1[3] = (int)(64u << 16);         // tensor_dim1[31:16]=0 | tile_dim0 @ bits 127:112
  g1[4] = (int)16u;                 // tile_dim1 @ bits 143:128, tile_dim2=0
  g1[5] = (int)640u;                // tensor_dim0_stride[31:0]
  g1[6] = 0;                        // stride[47:32]=0 | tensor_dim1_stride lo = 0
  g1[7] = 0;
  tdm_v4i g2 = {0,0,0,0}, g3 = {0,0,0,0};         // 2D tile: groups 2/3 unused
  tdm_v8i g4 = {0,0,0,0,0,0,0,0};                 // extra group (6-arg builtin form), unused for 2D
  // Unprovable store: prevents GlobalOpt from classifying smem as never-written.
  if (__builtin_expect((uintptr_t)feat == 0, 0)) smem[0] = (_Float16)1.f;
  __builtin_amdgcn_tensor_load_to_lds(g0, g1, g2, g3, g4, 0);
  __builtin_amdgcn_s_wait_tensorcnt(0);
  // Escape the LDS object so the memory clobber may have written it -> loads stay live,
  // remain AS3 accesses (ds_load), and can be merged/scheduled freely.
  asm volatile("" :: "v"((void*)smem) : "memory");
  const _Float16* aslab = smem + wib * 16 * 64 + (lane & 15) * 64;
#else
  int mrow = tm*16 + (lane & 15);
  const _Float16* arow = rows + ((size_t)b*1920 + mrow)*FRW;
#endif

  v8f acc = {};
  #pragma unroll
  for (int ch = 0; ch < 2; ++ch) {
    int kloc = ch*32;
    v16h a, bm;
    #pragma unroll
    for (int h = 0; h < 16; ++h) {
      int kl = (h < 8) ? (h + hiA) : (h + 8 + hiA);
#if USE_TDM
      a[h] = aslab[kloc + kl];                    // ds_load from staged tile
#else
      a[h] = arow[kstart + kloc + kl];
#endif
      int kb = kstart + kloc + kl;
      bm[h] = ((kb >> 2) == xg) ? (_Float16)0.0625f : (_Float16)0.0f;
    }
    acc = __builtin_amdgcn_wmma_f32_16x16x32_f16(false, a, false, bm,
                                                 (short)0, acc, false, false);
  }
  int hiD = (lane >= 16) ? 8 : 0;                 // D layout: VGPR v holds M=v (+8 for hi lanes)
  #pragma unroll
  for (int v = 0; v < 8; ++v) {
    int mo  = tm*16 + v + hiD;
    int cch = mo / PH, yy = mo % PH;
    feat[((size_t)b*FC + 1 + cch)*NPTS + (size_t)yy*PW + xg] = acc[v];
  }
}

__global__ void k_coords(const float* obs, float* coord, float* feat, int t) {
  int i = blockIdx.x * blockDim.x + threadIdx.x;
  if (i >= BB*NPTS) return;
  int b = i / NPTS, n = i % NPTS;
  int row = n / PW, col = n % PW;
  float d = obs[((((size_t)b*TT + t)*CMAP + 3)*FRH + 4*(size_t)row)*FRW + 4*col];
  bool valid = (d > 50.f) && (d < 350.f);
  const float XC = (FRW-1)*0.5f, ZC = (FRH-1)*0.5f;
  float FOCAL = (FRW*0.5f) / tanf(39.5f * PIF / 180.f);
  float X  = (4.f*col - XC) * d / FOCAL;
  float Zp = ((float)(FRH-1 - 4*row) - ZC) * d / FOCAL;
  float px = X + 250.f;          // +VR*RES/2
  float py = d;
  float pz = Zp + 88.f;          // +CAM_H
  float cx = (px/5.f - 50.f)*0.02f;
  float cy = (py/5.f - 50.f)*0.02f;
  float cz = (pz/5.f - 32.f)*0.025f;
  if (!valid) { cx = 2.f; cy = 2.f; cz = 2.f; }
  coord[((size_t)b*3 + 0)*NPTS + n] = cx;
  coord[((size_t)b*3 + 1)*NPTS + n] = cy;
  coord[((size_t)b*3 + 2)*NPTS + n] = cz;
  feat[(size_t)b*FC*NPTS + n] = 1.f;   // ones channel
}

__global__ void k_splat(const float* coord, const float* feat, float* vox) {
  int i = blockIdx.x * blockDim.x + threadIdx.x;
  if (i >= BB*NPTS) return;
  int b = i / NPTS, n = i % NPTS;
  const int dims[3] = {VRN, VRN, ZBN};
  float pos[3], fl[3];
  #pragma unroll
  for (int d = 0; d < 3; ++d) {
    pos[d] = coord[((size_t)b*3 + d)*NPTS + n] * dims[d]*0.5f + dims[d]*0.5f;
    fl[d]  = floorf(pos[d]);
  }
  #pragma unroll
  for (int corner = 0; corner < 8; ++corner) {
    float idxf = 0.f, w = 1.f;
    #pragma unroll
    for (int d = 0; d < 3; ++d) {
      float p = fl[d] + (float)((corner >> (2-d)) & 1);
      float safe = (p > 0.f && p < (float)dims[d]) ? 1.f : 0.f;
      w *= (1.f - fabsf(pos[d] - p)) * safe;
      idxf = idxf*dims[d] + p*safe;
    }
    if (w != 0.f) {
      int idx = (int)idxf;
      for (int f = 0; f < FC; ++f) {
        float val = feat[((size_t)b*FC + f)*NPTS + n] * w;
        if (val != 0.f) atomicAdd(&vox[((size_t)b*FC + f)*GVOX + idx], val);
      }
    }
  }
}

__global__ void k_reduce(const float* vox, float* av) {
  int i = blockIdx.x * blockDim.x + threadIdx.x;
  if (i >= BB*FC*VRN*VRN) return;
  int xy = i % (VRN*VRN); int r = i / (VRN*VRN);
  int f = r % FC; int b = r / FC;
  const float* vp = vox + (size_t)i * ZBN;
  float agent = 0.f, allh = 0.f;
  for (int z = 0; z < ZBN; ++z) {
    float v = roundf(vp[z]);
    allh += v;
    if (z >= MINZ && z < MAXZ) agent += v;
  }
  int ii = xy / VRN, jj = xy % VRN;
  size_t chs = (size_t)LOCN*LOCN;
  size_t base = (size_t)b*CMAP*chs + (size_t)(240 + ii)*LOCN + (190 + jj);
  if (f == 0) {
    av[base + 0*chs] = fminf(fmaxf(agent, 0.f), 1.f);
    av[base + 1*chs] = fminf(fmaxf(allh , 0.f), 1.f);
  } else {
    av[base + (size_t)(3 + f)*chs] = fminf(fmaxf(agent*0.2f, 0.f), 1.f);
  }
}

__global__ void k_gsample(const float* __restrict__ src, float* __restrict__ dst,
                          const float* pose, int mode) {
  int i = blockIdx.x * blockDim.x + threadIdx.x;
  if (i >= BB*LOCN*LOCN) return;
  int b = i / (LOCN*LOCN); int p = i % (LOCN*LOCN);
  int r = p / LOCN, cO = p % LOCN;
  const float* lp = pose + b*3;
  float gx = -1.f + 2.f*cO/(LOCN-1);
  float gy = -1.f + 2.f*r /(LOCN-1);
  float px, py;
  if (mode == 0) {
    float th = (90.f - lp[2]) * (PIF/180.f);
    float ct = cosf(th), st = sinf(th);
    px = ct*gx - st*gy; py = st*gx + ct*gy;
  } else {
    float stx = -(lp[0]*20.f - 240.f)/240.f;
    float sty = -(lp[1]*20.f - 240.f)/240.f;
    px = gx + stx; py = gy + sty;
  }
  float x = (px + 1.f)*(LOCN-1)*0.5f;
  float y = (py + 1.f)*(LOCN-1)*0.5f;
  float x0 = floorf(x), y0 = floorf(y);
  int xi[4], yi[4]; float wt[4]; int k = 0;
  #pragma unroll
  for (int dy = 0; dy < 2; ++dy)
    #pragma unroll
    for (int dx = 0; dx < 2; ++dx) {
      float xf = x0 + dx, yf = y0 + dy;
      float w = (1.f - fabsf(x - xf)) * (1.f - fabsf(y - yf));
      bool ok = (xf >= 0.f) && (xf <= LOCN-1) && (yf >= 0.f) && (yf <= LOCN-1);
      int xc = (int)fminf(fmaxf(xf, 0.f), (float)(LOCN-1));
      int yc = (int)fminf(fmaxf(yf, 0.f), (float)(LOCN-1));
      xi[k] = xc; yi[k] = yc; wt[k] = ok ? w : 0.f; ++k;
    }
  size_t bbase = (size_t)b*CMAP*LOCN*LOCN;
  for (int ch = 0; ch < CMAP; ++ch) {
    const float* im = src + bbase + (size_t)ch*LOCN*LOCN;
    __builtin_prefetch(im + (size_t)yi[0]*LOCN + xi[0], 0, 1);   // global_prefetch_b8
    float v = 0.f;
    #pragma unroll
    for (k = 0; k < 4; ++k) v += im[(size_t)yi[k]*LOCN + xi[k]] * wt[k];
    dst[bbase + (size_t)ch*LOCN*LOCN + (size_t)r*LOCN + cO] = v;
  }
}

__global__ void k_fuse(float* lmap, const float* trans, float* tmp0) {
  size_t i = (size_t)blockIdx.x * blockDim.x + threadIdx.x;
  size_t n = (size_t)BB*CMAP*LOCN*LOCN;
  if (i >= n) return;
  float v = fmaxf(lmap[i], trans[i]);
  lmap[i] = v;
  size_t chs = (size_t)LOCN*LOCN, per_b = (size_t)CMAP*chs;
  size_t b = i / per_b, rem = i % per_b;
  if (rem < chs) tmp0[b*chs + rem] = v;     // snapshot channel 0 for dilation
}

__global__ void k_dilate(float* lmap, const float* tmp0, const float* pose) {
  int i = blockIdx.x * blockDim.x + threadIdx.x;
  if (i >= BB*LOCN*LOCN) return;
  int b = i / (LOCN*LOCN); int p = i % (LOCN*LOCN);
  int r = p / LOCN, c = p % LOCN;
  const float* t0 = tmp0 + (size_t)b*LOCN*LOCN;
  float s = 0.f;
  for (int dy = -1; dy <= 1; ++dy)
    for (int dx = -1; dx <= 1; ++dx) {
      int rr = r + dy, cc = c + dx;
      if (rr >= 0 && rr < LOCN && cc >= 0 && cc < LOCN) s += t0[(size_t)rr*LOCN + cc];
    }
  size_t chs = (size_t)LOCN*LOCN;
  size_t base = (size_t)b*CMAP*chs + (size_t)r*LOCN + c;
  lmap[base + 0*chs] = (s > 0.5f) ? 1.f : 0.f;
  const float* lp = pose + b*3;
  int xg = (int)(lp[0]*20.f), yg = (int)(lp[1]*20.f);
  int dr = r - yg, dc = c - xg;
  lmap[base + 2*chs] = (abs(dr) <= 2 && abs(dc) <= 2) ? 1.f : 0.f;
  float disk = ((dr*dr + dc*dc) <= 1600) ? 1.f : 0.f;
  lmap[base + 3*chs] = fmaxf(lmap[base + 3*chs], disk);
}

__global__ void k_commit(const float* lmap, float* gmap, const int* lmb,
                         const unsigned char* updg, int t) {
  size_t i = (size_t)blockIdx.x * blockDim.x + threadIdx.x;
  size_t n = (size_t)BB*CMAP*LOCN*LOCN;
  if (i >= n) return;
  size_t per_b = (size_t)CMAP*LOCN*LOCN;
  int b = (int)(i / per_b);
  if (!updg[b*TT + t]) return;
  size_t rem = i % per_b;
  int ch = (int)(rem / ((size_t)LOCN*LOCN));
  int p = (int)(rem % ((size_t)LOCN*LOCN));
  int r = p / LOCN, c = p % LOCN;
  gmap[((size_t)b*CMAP + ch)*GLON*GLON + (size_t)(lmb[b*4] + r)*GLON + (lmb[b*4+2] + c)] = lmap[i];
}

__global__ void k_gupdate(float* pose, int* lmb, const unsigned char* updg, int t) {
  int b = threadIdx.x; if (b >= BB) return;
  if (!updg[b*TT + t]) return;
  float* lp = pose + b*3; float* gp = pose + 6 + b*3; float* og = pose + 12 + b*3;
  float gx = lp[0] + og[0], gy = lp[1] + og[1], go = lp[2] + og[2];
  int r = (int)(gy*20.f), c = (int)(gx*20.f);
  int gx1 = min(max(r - 240, 0), GLON - LOCN);
  int gy1 = min(max(c - 240, 0), GLON - LOCN);
  lmb[b*4+0] = gx1; lmb[b*4+1] = gx1 + LOCN;
  lmb[b*4+2] = gy1; lmb[b*4+3] = gy1 + LOCN;
  og[0] = gy1*0.05f; og[1] = gx1*0.05f; og[2] = 0.f;
  gp[0] = gx; gp[1] = gy; gp[2] = go;
  lp[0] = gx - og[0]; lp[1] = gy - og[1]; lp[2] = go - og[2];
}

__global__ void k_slice(float* lmap, const float* gmap, const int* lmb,
                        const unsigned char* updg, int t) {
  size_t i = (size_t)blockIdx.x * blockDim.x + threadIdx.x;
  size_t n = (size_t)BB*CMAP*LOCN*LOCN;
  if (i >= n) return;
  size_t per_b = (size_t)CMAP*LOCN*LOCN;
  int b = (int)(i / per_b);
  if (!updg[b*TT + t]) return;
  size_t rem = i % per_b;
  int ch = (int)(rem / ((size_t)LOCN*LOCN));
  int p = (int)(rem % ((size_t)LOCN*LOCN));
  int r = p / LOCN, c = p % LOCN;
  lmap[i] = gmap[((size_t)b*CMAP + ch)*GLON*GLON + (size_t)(lmb[b*4] + r)*GLON + (lmb[b*4+2] + c)];
}

__global__ void k_feats(const float* lmap, const float* gmap, float* out, int t) {
  size_t i = (size_t)blockIdx.x * blockDim.x + threadIdx.x;
  size_t n = (size_t)BB*24*LOCN*LOCN;
  if (i >= n) return;
  size_t per_b = (size_t)24*LOCN*LOCN;
  int b = (int)(i / per_b);
  size_t rem = i % per_b;
  int ch = (int)(rem / ((size_t)LOCN*LOCN));
  int p = (int)(rem % ((size_t)LOCN*LOCN));
  int r = p / LOCN, c = p % LOCN;
  size_t lchs = (size_t)LOCN*LOCN;
  float v;
  if (ch < 4) {
    v = lmap[((size_t)b*CMAP + ch)*lchs + (size_t)r*LOCN + c];
  } else if (ch < 8) {
    const float* g = gmap + ((size_t)b*CMAP + (ch - 4))*GLON*GLON;
    size_t r2 = (size_t)(2*r)*GLON + 2*c;
    v = fmaxf(fmaxf(g[r2], g[r2+1]), fmaxf(g[r2+GLON], g[r2+GLON+1]));
  } else {
    v = lmap[((size_t)b*CMAP + (ch - 4))*lchs + (size_t)r*LOCN + c];
  }
  out[OFF_FE + (((size_t)b*TT + t)*24 + ch)*lchs + (size_t)r*LOCN + c] = v;
}

__global__ void k_outpose(const float* pose, const int* lmb, float* out, int t) {
  int b = threadIdx.x; if (b >= BB) return;
  const float* lp = pose + b*3; const float* gp = pose + 6 + b*3; const float* og = pose + 12 + b*3;
  size_t bt = (size_t)b*TT + t;
  for (int k = 0; k < 3; ++k) {
    out[OFF_LP + bt*3 + k] = lp[k];
    out[OFF_GP + bt*3 + k] = gp[k];
    out[OFF_OG + bt*3 + k] = og[k];
  }
  for (int k = 0; k < 4; ++k) out[OFF_LB + bt*4 + k] = (float)lmb[b*4 + k];
}

extern "C" void kernel_launch(void* const* d_in, const int* in_sizes, int n_in,
                              void* d_out, int out_size, void* d_ws, size_t ws_size,
                              hipStream_t stream) {
  const float* obs            = (const float*)d_in[0];
  const float* pdelta         = (const float*)d_in[1];
  const unsigned char* dones  = (const unsigned char*)d_in[2];
  const unsigned char* updg   = (const unsigned char*)d_in[3];
  float* ws  = (float*)d_ws;
  float* out = (float*)d_out;

  float* lmap  = ws + WS_LMAP;
  float* gmap  = ws + WS_GMAP;
  float* av    = ws + WS_AV;
  float* rot   = ws + WS_ROT;
  float* vox   = ws + WS_VOX;
  float* feat  = ws + WS_FEAT;
  float* coord = ws + WS_COORD;
  float* tmp0  = ws + WS_TMP0;
  _Float16* rows = (_Float16*)(ws + WS_ROWS);
  float* pose  = ws + WS_POSE;
  int*   lmb   = (int*)(ws + WS_LMB);

  // re-initialize carried state from inputs every call (deterministic)
  (void)hipMemcpyAsync(lmap, d_in[5], (size_t)BB*CMAP*LOCN*LOCN*sizeof(float), hipMemcpyDeviceToDevice, stream);
  (void)hipMemcpyAsync(gmap, d_in[6], (size_t)BB*CMAP*GLON*GLON*sizeof(float), hipMemcpyDeviceToDevice, stream);
  (void)hipMemcpyAsync(pose,      d_in[7],  BB*3*sizeof(float), hipMemcpyDeviceToDevice, stream);
  (void)hipMemcpyAsync(pose + 6,  d_in[8],  BB*3*sizeof(float), hipMemcpyDeviceToDevice, stream);
  (void)hipMemcpyAsync(pose + 12, d_in[10], BB*3*sizeof(float), hipMemcpyDeviceToDevice, stream);
  (void)hipMemcpyAsync(lmb,       d_in[9],  BB*4*sizeof(int),   hipMemcpyDeviceToDevice, stream);

  dim3 blk(256);
  auto nb = [](size_t n) { return dim3((unsigned)((n + 255) / 256)); };

  for (int t = 0; t < TT; ++t) {
    k_apply_done_map<<<nb((size_t)BB*CMAP*LOCN*LOCN), blk, 0, stream>>>(lmap, (size_t)CMAP*LOCN*LOCN, dones, t);
    k_apply_done_map<<<nb((size_t)BB*CMAP*GLON*GLON), blk, 0, stream>>>(gmap, (size_t)CMAP*GLON*GLON, dones, t);
    k_reset_pose<<<1, BB, 0, stream>>>(pose, lmb, dones, t);
    k_new_pose  <<<1, BB, 0, stream>>>(pose, pdelta, t);

    k_rowsum<<<nb((size_t)BB*NSEM*PH*FRW), blk, 0, stream>>>(obs, rows, t);
    k_pool_wmma<<<(BB*120*10*32)/256, 256, 0, stream>>>(rows, feat);   // 300 blocks, EXEC all-ones
    k_coords<<<nb((size_t)BB*NPTS), blk, 0, stream>>>(obs, coord, feat, t);

    k_zero<<<2048, blk, 0, stream>>>(vox, (size_t)BB*FC*GVOX);
    k_splat<<<nb((size_t)BB*NPTS), blk, 0, stream>>>(coord, feat, vox);
    k_zero<<<2048, blk, 0, stream>>>(av, (size_t)BB*CMAP*LOCN*LOCN);
    k_reduce<<<nb((size_t)BB*FC*VRN*VRN), blk, 0, stream>>>(vox, av);

    k_gsample<<<nb((size_t)BB*LOCN*LOCN), blk, 0, stream>>>(av, rot, pose, 0);   // rotate
    k_gsample<<<nb((size_t)BB*LOCN*LOCN), blk, 0, stream>>>(rot, av, pose, 1);   // translate
    k_fuse<<<nb((size_t)BB*CMAP*LOCN*LOCN), blk, 0, stream>>>(lmap, av, tmp0);
    k_dilate<<<nb((size_t)BB*LOCN*LOCN), blk, 0, stream>>>(lmap, tmp0, pose);

    k_commit<<<nb((size_t)BB*CMAP*LOCN*LOCN), blk, 0, stream>>>(lmap, gmap, lmb, updg, t);
    k_gupdate<<<1, BB, 0, stream>>>(pose, lmb, updg, t);
    k_slice<<<nb((size_t)BB*CMAP*LOCN*LOCN), blk, 0, stream>>>(lmap, gmap, lmb, updg, t);

    k_feats<<<nb((size_t)BB*24*LOCN*LOCN), blk, 0, stream>>>(lmap, gmap, out, t);
    k_outpose<<<1, BB, 0, stream>>>(pose, lmb, out, t);
  }

  (void)hipMemcpyAsync(out + OFF_LMAP, lmap, (size_t)BB*CMAP*LOCN*LOCN*sizeof(float), hipMemcpyDeviceToDevice, stream);
  (void)hipMemcpyAsync(out + OFF_GMAP, gmap, (size_t)BB*CMAP*GLON*GLON*sizeof(float), hipMemcpyDeviceToDevice, stream);
}
